// SGATHAConv_25778393711291
// MI455X (gfx1250) — compile-verified, matrix operand order checked by reading, not run on hardware
//
#include <hip/hip_runtime.h>

#define IN_FEATS 128
#define OUT_FEATS 64
#define NEG_SLOPE 0.2f

typedef __attribute__((ext_vector_type(16))) __bf16 v16bf;
typedef __attribute__((ext_vector_type(8)))  __bf16 v8bf;
typedef __attribute__((ext_vector_type(8)))  float  v8f;

__device__ __forceinline__ __bf16 f2bf(float f) {
    unsigned u = __builtin_bit_cast(unsigned, f);
    unsigned r = (u + 0x7FFFu + ((u >> 16) & 1u)) >> 16;   // round-to-nearest-even
    unsigned short s = (unsigned short)r;
    return __builtin_bit_cast(__bf16, s);
}

__device__ __forceinline__ float leaky(float e) {
    return e > 0.0f ? e : NEG_SLOPE * e;
}

// ---------------- one-shot f32 -> bf16 conversion (bandwidth pass) ----------
__global__ void cvt_f32_bf16(const float* __restrict__ in, __bf16* __restrict__ outp,
                             int n) {
    int i = blockIdx.x * blockDim.x + threadIdx.x;
    if (i < n) outp[i] = f2bf(in[i]);
}

// ---------------- GEMM: h[N,64] = feat[N,128] @ W[64,128]^T via bf16 WMMA ----
// block = 128 threads (4 waves). Wave w computes 16 nodes x outfeats [16w,16w+16).
// A fragment (16x32 bf16): lane l (half=l>>4, m=l&15) holds row m,
//   K = {kc+half*8 .. +7} in a[0:7] and {kc+16+half*8 .. +7} in a[8:15].
__global__ void gat_gemm_wmma(const __bf16* __restrict__ featb,
                              const __bf16* __restrict__ Wb,
                              float* __restrict__ h, int n_nodes) {
    const int lane  = threadIdx.x & 31;
    const int wtile = threadIdx.x >> 5;      // 0..3 : out-feature tile
    const int half  = lane >> 4;             // 0/1
    const int m16   = lane & 15;
    const int row   = blockIdx.x * 16 + m16;
    const int rowc  = row < n_nodes ? row : n_nodes - 1;   // clamp: EXEC stays all-1
    const int ncol  = wtile * 16 + m16;

    const __bf16* fa = featb + (size_t)rowc * IN_FEATS + half * 8;
    const __bf16* fb = Wb    + (size_t)ncol * IN_FEATS + half * 8;

    v8f c = {};
#pragma unroll
    for (int kc = 0; kc < IN_FEATS; kc += 32) {
        v8bf a0 = *(const v8bf*)(fa + kc);        // 16B aligned -> global_load_b128
        v8bf a1 = *(const v8bf*)(fa + kc + 16);
        v8bf b0 = *(const v8bf*)(fb + kc);
        v8bf b1 = *(const v8bf*)(fb + kc + 16);
        v16bf a, b;
#pragma unroll
        for (int j = 0; j < 8; ++j) {
            a[j] = a0[j]; a[8 + j] = a1[j];
            b[j] = b0[j]; b[8 + j] = b1[j];
        }
        c = __builtin_amdgcn_wmma_f32_16x16x32_bf16(
                false, a, false, b, (short)0, c, false, false);
    }
    // D layout: VGPR r -> M = r + 8*half, N = m16.
    // Lane's 8 stores stride OUT_FEATS floats (256 B) -> immediate offsets.
    const int base_row = blockIdx.x * 16 + 8 * half;
    float* hp = h + (size_t)base_row * OUT_FEATS + wtile * 16 + m16;
    if (blockIdx.x * 16 + 16 <= n_nodes) {          // full tile: branch-free stores
#pragma unroll
        for (int r = 0; r < 8; ++r)
            hp[(size_t)r * OUT_FEATS] = c[r];
    } else {                                        // ragged last block only
#pragma unroll
        for (int r = 0; r < 8; ++r)
            if (base_row + r < n_nodes)
                hp[(size_t)r * OUT_FEATS] = c[r];
    }
}

// ---------------- per-node attention projections ----------------------------
__global__ void gat_el_er(const float* __restrict__ h,
                          const float* __restrict__ al,
                          const float* __restrict__ ar,
                          float* __restrict__ el, float* __restrict__ er, int n) {
    int i = blockIdx.x * blockDim.x + threadIdx.x;
    if (i >= n) return;
    const float* hp = h + (size_t)i * OUT_FEATS;
    float sl = 0.f, sr = 0.f;
#pragma unroll 8
    for (int f = 0; f < OUT_FEATS; ++f) { float v = hp[f]; sl += v * al[f]; sr += v * ar[f]; }
    el[i] = sl; er[i] = sr;
}

// ---------------- init / degrees / norms ------------------------------------
__global__ void gat_init_nodes(float* emax, float* denom, float* odeg, float* ideg, int n) {
    int i = blockIdx.x * blockDim.x + threadIdx.x;
    if (i >= n) return;
    emax[i] = -3.0e38f; denom[i] = 0.f; odeg[i] = 0.f; ideg[i] = 0.f;
}

__global__ void gat_deg(const int* __restrict__ src, const int* __restrict__ dst,
                        float* odeg, float* ideg, int E) {
    int e = blockIdx.x * blockDim.x + threadIdx.x;
    if (e >= E) return;
    atomicAdd(&odeg[src[e]], 1.0f);
    atomicAdd(&ideg[dst[e]], 1.0f);
}

__global__ void gat_norm(float* odeg, float* ideg, int n) {
    int i = blockIdx.x * blockDim.x + threadIdx.x;
    if (i >= n) return;
    odeg[i] = rsqrtf(fmaxf(odeg[i], 1.0f));  // out_norm = deg^-0.5
    ideg[i] = sqrtf(fmaxf(ideg[i], 1.0f));   // in_norm  = deg^+0.5
}

// ---------------- edge softmax ----------------------------------------------
__global__ void gat_emax(const float* __restrict__ el, const float* __restrict__ er,
                         const int* __restrict__ src, const int* __restrict__ dst,
                         float* emax, int E) {
    int e = blockIdx.x * blockDim.x + threadIdx.x;
    if (e >= E) return;
    float v = leaky(el[src[e]] + er[dst[e]]);
    atomicMax(&emax[dst[e]], v);             // global_atomic_max_num_f32
}

__global__ void gat_esum(const float* __restrict__ el, const float* __restrict__ er,
                         const int* __restrict__ src, const int* __restrict__ dst,
                         const float* __restrict__ emax,
                         float* __restrict__ ew, float* denom, int E) {
    int e = blockIdx.x * blockDim.x + threadIdx.x;
    if (e >= E) return;
    int d = dst[e];
    float ee = __expf(leaky(el[src[e]] + er[d]) - emax[d]);
    ew[e] = ee;
    atomicAdd(&denom[d], ee);
}

__global__ void gat_ewfin(float* __restrict__ ew, const float* __restrict__ denom,
                          const float* __restrict__ onorm,
                          const int* __restrict__ src, const int* __restrict__ dst, int E) {
    int e = blockIdx.x * blockDim.x + threadIdx.x;
    if (e >= E) return;
    ew[e] = ew[e] / denom[dst[e]] * onorm[src[e]];   // a[e] * out_norm[src]
}

// ---------------- hop propagation -------------------------------------------
__global__ void zero_f(float* __restrict__ p, int n) {
    int i = blockIdx.x * blockDim.x + threadIdx.x;
    if (i < n) p[i] = 0.f;
}

// 64 consecutive threads handle one edge's 64 features (coalesced)
__global__ void gat_scatter(const float* __restrict__ cur, float* __restrict__ nxt,
                            const float* __restrict__ ew,
                            const int* __restrict__ src, const int* __restrict__ dst,
                            int total) {
    int idx = blockIdx.x * blockDim.x + threadIdx.x;
    if (idx >= total) return;
    int e = idx >> 6;
    int f = idx & 63;
    int s = src[e], d = dst[e];
    atomicAdd(&nxt[d * OUT_FEATS + f], ew[e] * cur[s * OUT_FEATS + f]);
}

__global__ void gat_scale(float* __restrict__ p, const float* __restrict__ inorm,
                          int total) {
    int i = blockIdx.x * blockDim.x + threadIdx.x;
    if (i < total) p[i] *= inorm[i >> 6];
}

// ---------------- driver -----------------------------------------------------
extern "C" void kernel_launch(void* const* d_in, const int* in_sizes, int n_in,
                              void* d_out, int out_size, void* d_ws, size_t ws_size,
                              hipStream_t stream) {
    const float* feat = (const float*)d_in[0];
    const float* W    = (const float*)d_in[1];
    const float* al   = (const float*)d_in[2];
    const float* ar   = (const float*)d_in[3];
    const int*   src  = (const int*)d_in[4];
    const int*   dst  = (const int*)d_in[5];
    const int N = in_sizes[0] / IN_FEATS;
    const int E = in_sizes[4];

    char* ws = (char*)d_ws;
    size_t off = 0;
    auto alloc = [&](size_t bytes) -> void* {
        void* p = (void*)(ws + off);
        off += (bytes + 255) & ~(size_t)255;
        return p;
    };
    float*  h     = (float*)alloc((size_t)N * OUT_FEATS * sizeof(float));
    float*  tmp   = (float*)alloc((size_t)N * OUT_FEATS * sizeof(float));
    __bf16* featb = (__bf16*)alloc((size_t)N * IN_FEATS * sizeof(__bf16));
    __bf16* Wb    = (__bf16*)alloc((size_t)OUT_FEATS * IN_FEATS * sizeof(__bf16));
    float*  el    = (float*)alloc((size_t)N * sizeof(float));
    float*  er    = (float*)alloc((size_t)N * sizeof(float));
    float*  emax  = (float*)alloc((size_t)N * sizeof(float));
    float*  denom = (float*)alloc((size_t)N * sizeof(float));
    float*  onorm = (float*)alloc((size_t)N * sizeof(float));
    float*  inorm = (float*)alloc((size_t)N * sizeof(float));
    float*  ew    = (float*)alloc((size_t)E * sizeof(float));
    float*  out   = (float*)d_out;

    const int TB = 256;
    const int nbN = (N + TB - 1) / TB;
    const int nbE = (E + TB - 1) / TB;
    const int NF  = N * OUT_FEATS;          // 6.4M  (fits int32)
    const int NK  = N * IN_FEATS;           // 12.8M (fits int32)
    const int WK  = OUT_FEATS * IN_FEATS;
    const int EF  = E * OUT_FEATS;          // 102.4M (fits int32)
    const int nbNF = (NF + TB - 1) / TB;
    const int nbNK = (NK + TB - 1) / TB;
    const int nbWK = (WK + TB - 1) / TB;
    const int nbEF = (EF + TB - 1) / TB;

    gat_init_nodes<<<nbN, TB, 0, stream>>>(emax, denom, onorm, inorm, N);
    cvt_f32_bf16<<<nbNK, TB, 0, stream>>>(feat, featb, NK);
    cvt_f32_bf16<<<nbWK, TB, 0, stream>>>(W, Wb, WK);
    gat_gemm_wmma<<<(N + 15) / 16, 128, 0, stream>>>(featb, Wb, h, N);
    gat_el_er<<<nbN, TB, 0, stream>>>(h, al, ar, el, er, N);
    gat_deg<<<nbE, TB, 0, stream>>>(src, dst, onorm, inorm, E);
    gat_norm<<<nbN, TB, 0, stream>>>(onorm, inorm, N);
    gat_emax<<<nbE, TB, 0, stream>>>(el, er, src, dst, emax, E);
    gat_esum<<<nbE, TB, 0, stream>>>(el, er, src, dst, emax, ew, denom, E);
    gat_ewfin<<<nbE, TB, 0, stream>>>(ew, denom, onorm, src, dst, E);

    // hop 1: h -> tmp
    zero_f<<<nbNF, TB, 0, stream>>>(tmp, NF);
    gat_scatter<<<nbEF, TB, 0, stream>>>(h, tmp, ew, src, dst, EF);
    gat_scale<<<nbNF, TB, 0, stream>>>(tmp, inorm, NF);
    // hop 2: tmp -> h (reuse)
    zero_f<<<nbNF, TB, 0, stream>>>(h, NF);
    gat_scatter<<<nbEF, TB, 0, stream>>>(tmp, h, ew, src, dst, EF);
    gat_scale<<<nbNF, TB, 0, stream>>>(h, inorm, NF);
    // hop 3: h -> d_out
    zero_f<<<nbNF, TB, 0, stream>>>(out, NF);
    gat_scatter<<<nbEF, TB, 0, stream>>>(h, out, ew, src, dst, EF);
    gat_scale<<<nbNF, TB, 0, stream>>>(out, inorm, NF);
}
